// MSSharePixelDecoder_60730837566127
// MI455X (gfx1250) — matrix operand order repaired and checked by reading, not compiled, not run on hardware
//
#include <hip/hip_runtime.h>

typedef __attribute__((ext_vector_type(8)))  float          v8f;
typedef __attribute__((ext_vector_type(16))) __bf16         v16bf;
typedef __attribute__((ext_vector_type(8)))  unsigned short v8us;

union Frag { v8us h[2]; v16bf v; };

#define B_    2
#define D_    256
#define LEN_  12096
#define NH_   8
#define HD_   32
#define TWO_PI_ 6.28318530717958647692f

__device__ __forceinline__ unsigned short f2bf(float f) {
  unsigned int u = __float_as_uint(f);
  u += 0x7FFFu + ((u >> 16) & 1u);           // round-to-nearest-even
  return (unsigned short)(u >> 16);
}

__device__ __forceinline__ float wsum(float v) {
#pragma unroll
  for (int o = 16; o > 0; o >>= 1) v += __shfl_xor(v, o, 32);
  return v;
}

__device__ __forceinline__ void level_of(int q, int& lvl, int& hw, int& p0) {
  if (q < 576)        { lvl = 0; hw = 24; p0 = 0; }
  else if (q < 2880)  { lvl = 1; hw = 48; p0 = 576; }
  else                { lvl = 2; hw = 96; p0 = 2880; }
}

// ---------------------------------------------------------------------------
// Generic WMMA GEMM: C[M,N] = A[M,K](bf16) * Bt[N,K]^T(bf16) + bias, wave32.
// One wave computes a 16 x (16*NT) tile using v_wmma_f32_16x16x32_bf16.
// ---------------------------------------------------------------------------
template<int NT>
__global__ void __launch_bounds__(256) wmma_gemm_kernel(
    const unsigned short* __restrict__ A,
    const unsigned short* __restrict__ Bt,
    const float* __restrict__ bias,
    float* __restrict__ Cf,
    unsigned short* __restrict__ Cb,
    int M, int N, int K, int relu, int total_tiles)
{
  int wid = blockIdx.x * 8 + (threadIdx.x >> 5);
  if (wid >= total_tiles) return;
  int lane = threadIdx.x & 31;
  int hi = lane >> 4, lo = lane & 15;
  int ntn = N / (16 * NT);
  int tm = wid / ntn, tn = wid % ntn;

  const unsigned short* Arow = A + (size_t)(tm * 16 + lo) * K;
  const unsigned short* Brow[NT];
#pragma unroll
  for (int t = 0; t < NT; ++t)
    Brow[t] = Bt + (size_t)(tn * (16 * NT) + t * 16 + lo) * K;

  v8f zero8;
#pragma unroll
  for (int i = 0; i < 8; ++i) zero8[i] = 0.0f;
  v8f acc[NT];
#pragma unroll
  for (int t = 0; t < NT; ++t) acc[t] = zero8;

  for (int k0 = 0; k0 < K; k0 += 32) {
    if (k0 + 32 < K) __builtin_prefetch((const void*)(Arow + k0 + 32), 0, 0);
    Frag a;                                   // A 16x32 bf16 per-lane layout
    a.h[0] = *(const v8us*)(Arow + k0 + hi * 8);
    a.h[1] = *(const v8us*)(Arow + k0 + 16 + hi * 8);
#pragma unroll
    for (int t = 0; t < NT; ++t) {
      Frag b;                                 // B 32x16 bf16 per-lane layout
      b.h[0] = *(const v8us*)(Brow[t] + k0 + hi * 8);
      b.h[1] = *(const v8us*)(Brow[t] + k0 + 16 + hi * 8);
      acc[t] = __builtin_amdgcn_wmma_f32_16x16x32_bf16(
          false, a.v, false, b.v, (short)0, acc[t], false, false);
    }
  }

#pragma unroll
  for (int t = 0; t < NT; ++t) {
    int n = tn * (16 * NT) + t * 16 + lo;
    float bv = bias ? bias[n] : 0.0f;
#pragma unroll
    for (int r = 0; r < 8; ++r) {
      int m = tm * 16 + hi * 8 + r;           // C/D layout: VGPR r -> M=r(+8)
      float v = acc[t][r] + bv;
      if (relu) v = fmaxf(v, 0.0f);
      size_t idx = (size_t)m * N + n;
      if (Cf) Cf[idx] = v;
      if (Cb) Cb[idx] = f2bf(v);
    }
  }
}

// ---------------------------------------------------------------------------
// 3x3 'SAME' conv as implicit-im2col WMMA GEMM.
// ybf: [B,192,192,256] bf16 (NHWC). Bt: [256, 2304] bf16 (kk = tap*256 + c).
// Tap loop hoisted: dh/dw/validity/base pointer computed once per tap, the
// inner 8 K-steps are pure load+WMMA.
// ---------------------------------------------------------------------------
__global__ void __launch_bounds__(256) conv3x3_wmma_kernel(
    const unsigned short* __restrict__ ybf,
    const unsigned short* __restrict__ Bt,
    float* __restrict__ Cf, int total_tiles)
{
  const int Kc = 2304, HWc = 192 * 192;
  int wid = blockIdx.x * 8 + (threadIdx.x >> 5);
  if (wid >= total_tiles) return;
  int lane = threadIdx.x & 31;
  int hi = lane >> 4, lo = lane & 15;
  int tm = wid >> 2, tn = wid & 3;            // N=256 -> 4 tiles of 64

  int m = tm * 16 + lo;
  int b = m / HWc, p = m % HWc;
  int h = p / 192, w = p % 192;

  const unsigned short* Brow[4];
#pragma unroll
  for (int t = 0; t < 4; ++t)
    Brow[t] = Bt + (size_t)(tn * 64 + t * 16 + lo) * Kc;

  v8us z8;
#pragma unroll
  for (int i = 0; i < 8; ++i) z8[i] = 0;
  v8f zf;
#pragma unroll
  for (int i = 0; i < 8; ++i) zf[i] = 0.0f;
  v8f acc[4];
#pragma unroll
  for (int t = 0; t < 4; ++t) acc[t] = zf;

  int ktap = 0;
  for (int dh = -1; dh <= 1; ++dh) {
    int hs = h + dh;
    bool vh = (unsigned)hs < 192u;
    size_t rowbase = (size_t)(b * 192 + hs) * 192;
    for (int dw = -1; dw <= 1; ++dw, ktap += 256) {
      int ws2 = w + dw;
      bool valid = vh && ((unsigned)ws2 < 192u);
      const unsigned short* abase = ybf + (rowbase + ws2) * 256;
#pragma unroll 2
      for (int kc = 0; kc < 256; kc += 32) {
        Frag a;
        if (valid) {
          a.h[0] = *(const v8us*)(abase + kc + hi * 8);
          a.h[1] = *(const v8us*)(abase + kc + 16 + hi * 8);
        } else { a.h[0] = z8; a.h[1] = z8; }
#pragma unroll
        for (int t = 0; t < 4; ++t) {
          Frag bfr;
          const unsigned short* br = Brow[t] + ktap + kc;
          bfr.h[0] = *(const v8us*)(br + hi * 8);
          bfr.h[1] = *(const v8us*)(br + 16 + hi * 8);
          acc[t] = __builtin_amdgcn_wmma_f32_16x16x32_bf16(
              false, a.v, false, bfr.v, (short)0, acc[t], false, false);
        }
      }
    }
  }
#pragma unroll
  for (int t = 0; t < 4; ++t) {
    int n = tn * 64 + t * 16 + lo;
#pragma unroll
    for (int r = 0; r < 8; ++r) {
      int mm = tm * 16 + hi * 8 + r;
      Cf[(size_t)mm * 256 + n] = acc[t][r];
    }
  }
}

// ---------------------------------------------------------------------------
// Elementwise / data-movement kernels
// ---------------------------------------------------------------------------
__global__ void wT_kernel(const float* __restrict__ in, unsigned short* __restrict__ out,
                          int K, int N) {           // [K,N] f32 -> [N,K] bf16
  int t = blockIdx.x * 256 + threadIdx.x;
  if (t >= K * N) return;
  int k = t % K, n = t / K;
  out[t] = f2bf(in[(size_t)k * N + n]);
}

__global__ void wC_kernel(const float* __restrict__ in, unsigned short* __restrict__ out,
                          int total) {              // straight f32 -> bf16
  int t = blockIdx.x * 256 + threadIdx.x;
  if (t >= total) return;
  out[t] = f2bf(in[t]);
}

__global__ void fpnT_kernel(const float* __restrict__ w, unsigned short* __restrict__ out) {
  int t = blockIdx.x * 256 + threadIdx.x;           // OIHW -> [O, tap*256+c]
  if (t >= 256 * 2304) return;
  int kk = t % 2304, o = t / 2304;
  int tap = kk >> 8, c = kk & 255;
  int dh = tap / 3, dw = tap % 3;
  out[t] = f2bf(w[(((size_t)o * 256 + c) * 3 + dh) * 3 + dw]);
}

__global__ void feat2rm_kernel(const float* __restrict__ in, unsigned short* __restrict__ out,
                               int C, int HW, int total) {   // [B,C,HW] -> [B*HW,C] bf16
  int t = blockIdx.x * 256 + threadIdx.x;
  if (t >= total) return;
  int c = t % C, p = (t / C) % HW, b = t / (C * HW);
  out[t] = f2bf(in[((size_t)b * C + c) * HW + p]);
}

__global__ void pos_embed_kernel(const float* __restrict__ level_embed, float* __restrict__ pos) {
  int t = blockIdx.x * 256 + threadIdx.x;
  if (t >= LEN_ * 256) return;
  int d = t & 255, q = t >> 8;
  int lvl, hw, p0; level_of(q, lvl, hw, p0);
  int p = q - p0, row = p / hw, col = p % hw;
  float coord = (d < 128) ? ((row + 1) / (hw + 1e-6f)) * TWO_PI_
                          : ((col + 1) / (hw + 1e-6f)) * TWO_PI_;
  int i = (d < 128) ? d : d - 128;
  float dt = powf(10000.0f, (2.0f * (i >> 1)) / 128.0f);
  float v = coord / dt;
  float pe = (i & 1) ? cosf(v) : sinf(v);
  pos[t] = pe + level_embed[lvl * 256 + d];
}

__global__ void addpos_kernel(const float* __restrict__ src, const float* __restrict__ pos,
                              unsigned short* __restrict__ qb, int total) {
  int t = blockIdx.x * 256 + threadIdx.x;
  if (t >= total) return;
  qb[t] = f2bf(src[t] + pos[t % (LEN_ * 256)]);
}

__global__ void softmax12_kernel(float* __restrict__ aw, int total) {
  int t = blockIdx.x * 256 + threadIdx.x;
  if (t >= total) return;
  float* a = aw + (size_t)t * 12;
  float mx = a[0];
#pragma unroll
  for (int i = 1; i < 12; ++i) mx = fmaxf(mx, a[i]);
  float s = 0.0f, e[12];
#pragma unroll
  for (int i = 0; i < 12; ++i) { e[i] = expf(a[i] - mx); s += e[i]; }
  float inv = 1.0f / s;
#pragma unroll
  for (int i = 0; i < 12; ++i) a[i] = e[i] * inv;
}

// one wave per (b,q,head); lane = channel within HD=32
__global__ void __launch_bounds__(256) msdeform_kernel(
    const float* __restrict__ value, const float* __restrict__ off,
    const float* __restrict__ aw, unsigned short* __restrict__ out_bf)
{
  int wid = blockIdx.x * 8 + (threadIdx.x >> 5);
  const int total = B_ * LEN_ * NH_;
  if (wid >= total) return;
  int lane = threadIdx.x & 31;
  int h = wid & 7, q = (wid >> 3) % LEN_, b = wid / (8 * LEN_);

  int qlvl, qhw, qp0; level_of(q, qlvl, qhw, qp0);
  int qp = q - qp0;
  float refx = ((qp % qhw) + 0.5f) / qhw;
  float refy = ((qp / qhw) + 0.5f) / qhw;

  const int WH[3] = {24, 48, 96};
  const int P0[3] = {0, 576, 2880};
  size_t obase = (size_t)wid * 24;
  size_t abase = (size_t)wid * 12;
  float acc = 0.0f;

  for (int lvl = 0; lvl < 3; ++lvl) {
    int Wl = WH[lvl], p0 = P0[lvl];
    float invW = 1.0f / Wl;
#pragma unroll
    for (int pt = 0; pt < 4; ++pt) {
      float ox = off[obase + (lvl * 4 + pt) * 2 + 0];
      float oy = off[obase + (lvl * 4 + pt) * 2 + 1];
      float wgt = aw[abase + lvl * 4 + pt];
      float x = (refx + ox * invW) * Wl - 0.5f;
      float y = (refy + oy * invW) * Wl - 0.5f;
      float xf = floorf(x), yf = floorf(y);
      float wx = x - xf, wy = y - yf;
      int x0 = (int)xf, y0 = (int)yf;
      float s = 0.0f;
#pragma unroll
      for (int dy = 0; dy < 2; ++dy) {
#pragma unroll
        for (int dx = 0; dx < 2; ++dx) {
          int xx = x0 + dx, yy = y0 + dy;
          float ww = (dx ? wx : 1.0f - wx) * (dy ? wy : 1.0f - wy);
          if ((unsigned)xx < (unsigned)Wl && (unsigned)yy < (unsigned)Wl) {
            s += ww * value[(((size_t)b * LEN_ + p0 + yy * Wl + xx) * 8 + h) * 32 + lane];
          }
        }
      }
      acc += s * wgt;
    }
  }
  out_bf[((size_t)(b * LEN_ + q)) * 256 + h * 32 + lane] = f2bf(acc);
}

// one wave per token: src = LN(src + delta); also emits bf16 copy
__global__ void __launch_bounds__(256) resln_kernel(
    float* __restrict__ src, const float* __restrict__ delta,
    const float* __restrict__ s, const float* __restrict__ bpar,
    unsigned short* __restrict__ src_bf, int rows)
{
  int row = blockIdx.x * 8 + (threadIdx.x >> 5);
  if (row >= rows) return;
  int lane = threadIdx.x & 31;
  float x[8], sum = 0.0f;
#pragma unroll
  for (int i = 0; i < 8; ++i) {
    int c = lane + 32 * i;
    x[i] = src[(size_t)row * 256 + c] + delta[(size_t)row * 256 + c];
    sum += x[i];
  }
  float mean = wsum(sum) * (1.0f / 256.0f);
  float vs = 0.0f;
#pragma unroll
  for (int i = 0; i < 8; ++i) { float d = x[i] - mean; vs += d * d; }
  float inv = rsqrtf(wsum(vs) * (1.0f / 256.0f) + 1e-5f);
#pragma unroll
  for (int i = 0; i < 8; ++i) {
    int c = lane + 32 * i;
    float o = (x[i] - mean) * inv * s[c] + bpar[c];
    src[(size_t)row * 256 + c] = o;
    src_bf[(size_t)row * 256 + c] = f2bf(o);
  }
}

// GroupNorm(32 groups) over [B,HW,256]; mode 0: ->src(+bf16) at qoff,
// mode 1: ->f32 buffer [B,HW,256], mode 2: relu -> CHW f32 output
__global__ void __launch_bounds__(256) groupnorm_kernel(
    const float* __restrict__ x, const float* __restrict__ gs, const float* __restrict__ gb,
    int HW, int mode, int qoff,
    float* __restrict__ o_src, unsigned short* __restrict__ o_srcbf,
    float* __restrict__ o_f32, float* __restrict__ o_chw)
{
  __shared__ float sh1[256], sh2[256];
  __shared__ float s_mean, s_inv;
  int b = blockIdx.x >> 5, g = blockIdx.x & 31;
  int tid = threadIdx.x;
  int n = HW * 8;
  float s1 = 0.0f, s2 = 0.0f;
  for (int j = tid; j < n; j += 256) {
    int p = j >> 3, c = (g << 3) + (j & 7);
    float v = x[((size_t)b * HW + p) * 256 + c];
    s1 += v; s2 += v * v;
  }
  sh1[tid] = s1; sh2[tid] = s2;
  __syncthreads();
  for (int st = 128; st > 0; st >>= 1) {
    if (tid < st) { sh1[tid] += sh1[tid + st]; sh2[tid] += sh2[tid + st]; }
    __syncthreads();
  }
  if (tid == 0) {
    float mean = sh1[0] / n;
    float var = sh2[0] / n - mean * mean;
    s_mean = mean; s_inv = rsqrtf(var + 1e-5f);
  }
  __syncthreads();
  float mean = s_mean, inv = s_inv;
  for (int j = tid; j < n; j += 256) {
    int p = j >> 3, c = (g << 3) + (j & 7);
    float v = (x[((size_t)b * HW + p) * 256 + c] - mean) * inv * gs[c] + gb[c];
    if (mode == 0) {
      size_t o = ((size_t)b * LEN_ + qoff + p) * 256 + c;
      o_src[o] = v; o_srcbf[o] = f2bf(v);
    } else if (mode == 1) {
      o_f32[((size_t)b * HW + p) * 256 + c] = v;
    } else {
      o_chw[((size_t)b * 256 + c) * HW + p] = fmaxf(v, 0.0f);
    }
  }
}

// y_bf[B,192,192,256] = lat + bilinear_up2x(src level-3 slice)
__global__ void upsample_add_kernel(const float* __restrict__ lat,
                                    const float* __restrict__ src,
                                    unsigned short* __restrict__ ybf, int total)
{
  int t = blockIdx.x * 256 + threadIdx.x;
  if (t >= total) return;
  int c = t & 255;
  int idx = t >> 8;
  int wo = idx % 192, ho = (idx / 192) % 192, b = idx / (192 * 192);
  float ys = fmaxf((ho + 0.5f) * 0.5f - 0.5f, 0.0f);
  float xs = fmaxf((wo + 0.5f) * 0.5f - 0.5f, 0.0f);
  int y0 = (int)floorf(ys), x0 = (int)floorf(xs);
  float wy = ys - y0, wx = xs - x0;
  int y1 = min(y0 + 1, 95), x1 = min(x0 + 1, 95);
  auto at = [&](int yy, int xx) {
    return src[((size_t)b * LEN_ + 2880 + yy * 96 + xx) * 256 + c];
  };
  float v = at(y0, x0) * (1 - wy) * (1 - wx) + at(y0, x1) * (1 - wy) * wx
          + at(y1, x0) * wy * (1 - wx)       + at(y1, x1) * wy * wx;
  ybf[t] = f2bf(lat[t] + v);
}

__global__ void copyout_kernel(const float* __restrict__ src, float* __restrict__ out,
                               int qoff, int HW, int total)
{
  int t = blockIdx.x * 256 + threadIdx.x;
  if (t >= total) return;
  int p = t % HW, c = (t / HW) & 255, b = t / (HW * 256);
  out[t] = src[((size_t)b * LEN_ + qoff + p) * 256 + c];
}

// ---------------------------------------------------------------------------
extern "C" void kernel_launch(void* const* d_in, const int* in_sizes, int n_in,
                              void* d_out, int out_size, void* d_ws, size_t ws_size,
                              hipStream_t stream) {
  (void)in_sizes; (void)n_in; (void)out_size; (void)ws_size;
  const float* res2 = (const float*)d_in[0];
  const float* res3 = (const float*)d_in[1];
  const float* res4 = (const float*)d_in[2];
  const float* res5 = (const float*)d_in[3];
  const float* pw[3]  = {(const float*)d_in[4], (const float*)d_in[8],  (const float*)d_in[12]};
  const float* pb[3]  = {(const float*)d_in[5], (const float*)d_in[9],  (const float*)d_in[13]};
  const float* pgs[3] = {(const float*)d_in[6], (const float*)d_in[10], (const float*)d_in[14]};
  const float* pgb[3] = {(const float*)d_in[7], (const float*)d_in[11], (const float*)d_in[15]};
  const float* level_embed = (const float*)d_in[16];
  const float* off_w = (const float*)d_in[17];
  const float* off_b = (const float*)d_in[18];
  const float* aw_w  = (const float*)d_in[19];
  const float* aw_b  = (const float*)d_in[20];
  const float* val_w = (const float*)d_in[21];
  const float* val_b = (const float*)d_in[22];
  const float* out_w = (const float*)d_in[23];
  const float* out_b = (const float*)d_in[24];
  const float* ln1_s = (const float*)d_in[25];
  const float* ln1_b = (const float*)d_in[26];
  const float* ffn1_w = (const float*)d_in[27];
  const float* ffn1_b = (const float*)d_in[28];
  const float* ffn2_w = (const float*)d_in[29];
  const float* ffn2_b = (const float*)d_in[30];
  const float* ln2_s = (const float*)d_in[31];
  const float* ln2_b = (const float*)d_in[32];
  const float* lat_w  = (const float*)d_in[33];
  const float* lat_gs = (const float*)d_in[34];
  const float* lat_gb = (const float*)d_in[35];
  const float* fpn_w  = (const float*)d_in[36];
  const float* fpn_gs = (const float*)d_in[37];
  const float* fpn_gb = (const float*)d_in[38];
  float* out = (float*)d_out;

  // ---- workspace carve (bump allocator, 256B aligned) ----
  char* base = (char*)d_ws;
  size_t cur = 0;
  auto alloc = [&](size_t bytes) -> void* {
    void* p = base + cur; cur += (bytes + 255) & ~(size_t)255; return p;
  };
  const size_t TOK = (size_t)B_ * LEN_ * 256;          // 6,193,152
  float*          srcf   = (float*)alloc(TOK * 4);
  unsigned short* src_bf = (unsigned short*)alloc(TOK * 2);
  float*          pos    = (float*)alloc((size_t)LEN_ * 256 * 4);
  unsigned short* q_bf   = (unsigned short*)alloc(TOK * 2);
  float*          value  = (float*)alloc(TOK * 4);
  float*          offv   = (float*)alloc((size_t)B_ * LEN_ * 192 * 4);
  float*          awv    = (float*)alloc((size_t)B_ * LEN_ * 96 * 4);
  unsigned short* samp_bf = (unsigned short*)alloc(TOK * 2);
  float*          tmp256 = (float*)alloc(TOK * 4);
  unsigned short* hidden = (unsigned short*)alloc((size_t)B_ * LEN_ * 1024 * 2);
  unsigned short* AT     = (unsigned short*)alloc((size_t)2 * 36864 * 256 * 2); // also y_bf
  float*          Pbuf   = (float*)alloc((size_t)2 * 36864 * 256 * 4);          // proj/conv out
  float*          latb   = (float*)alloc((size_t)2 * 36864 * 256 * 4);
  unsigned short* val_wT = (unsigned short*)alloc((size_t)6 * 65536 * 2);
  unsigned short* off_wT = (unsigned short*)alloc((size_t)6 * 49152 * 2);
  unsigned short* aw_wT  = (unsigned short*)alloc((size_t)6 * 24576 * 2);
  unsigned short* out_wT = (unsigned short*)alloc((size_t)6 * 65536 * 2);
  unsigned short* f1T    = (unsigned short*)alloc((size_t)6 * 262144 * 2);
  unsigned short* f2T    = (unsigned short*)alloc((size_t)6 * 262144 * 2);
  unsigned short* pwc[3];
  pwc[0] = (unsigned short*)alloc((size_t)256 * 2048 * 2);
  pwc[1] = (unsigned short*)alloc((size_t)256 * 1024 * 2);
  pwc[2] = (unsigned short*)alloc((size_t)256 * 512 * 2);
  unsigned short* latc = (unsigned short*)alloc((size_t)256 * 256 * 2);
  unsigned short* fpnT = (unsigned short*)alloc((size_t)256 * 2304 * 2);

  auto gblk = [](size_t n) { return dim3((unsigned)((n + 255) / 256)); };

  auto gemm = [&](const unsigned short* A, const unsigned short* Bt, const float* bias,
                  float* Cf, unsigned short* Cb, int M, int N, int K, int relu) {
    if (N % 64 == 0) {
      int tiles = (M / 16) * (N / 64);
      wmma_gemm_kernel<4><<<dim3((tiles + 7) / 8), dim3(256), 0, stream>>>(
          A, Bt, bias, Cf, Cb, M, N, K, relu, tiles);
    } else {
      int tiles = (M / 16) * (N / 32);
      wmma_gemm_kernel<2><<<dim3((tiles + 7) / 8), dim3(256), 0, stream>>>(
          A, Bt, bias, Cf, Cb, M, N, K, relu, tiles);
    }
  };

  // ---- weight prep (f32 -> bf16, transpose to [N,K]) ----
  for (int l = 0; l < 6; ++l) {
    wT_kernel<<<gblk(65536), 256, 0, stream>>>(val_w + l * 65536, val_wT + l * 65536, 256, 256);
    wT_kernel<<<gblk(49152), 256, 0, stream>>>(off_w + l * 49152, off_wT + l * 49152, 256, 192);
    wT_kernel<<<gblk(24576), 256, 0, stream>>>(aw_w + l * 24576, aw_wT + l * 24576, 256, 96);
    wT_kernel<<<gblk(65536), 256, 0, stream>>>(out_w + l * 65536, out_wT + l * 65536, 256, 256);
    wT_kernel<<<gblk(262144), 256, 0, stream>>>(ffn1_w + l * 262144, f1T + l * 262144, 256, 1024);
    wT_kernel<<<gblk(262144), 256, 0, stream>>>(ffn2_w + l * 262144, f2T + l * 262144, 1024, 256);
  }
  wC_kernel<<<gblk(256 * 2048), 256, 0, stream>>>(pw[0], pwc[0], 256 * 2048); // [O,C] already
  wC_kernel<<<gblk(256 * 1024), 256, 0, stream>>>(pw[1], pwc[1], 256 * 1024);
  wC_kernel<<<gblk(256 * 512), 256, 0, stream>>>(pw[2], pwc[2], 256 * 512);
  wC_kernel<<<gblk(256 * 256), 256, 0, stream>>>(lat_w, latc, 256 * 256);
  fpnT_kernel<<<gblk(256 * 2304), 256, 0, stream>>>(fpn_w, fpnT);

  // ---- positional embedding (+ level embed) ----
  pos_embed_kernel<<<gblk((size_t)LEN_ * 256), 256, 0, stream>>>(level_embed, pos);

  // ---- input projections: conv1x1 + GN -> src ----
  const float* feats[3] = {res5, res4, res3};
  const int   fC[3] = {2048, 1024, 512};
  const int   fHW[3] = {576, 2304, 9216};
  const int   qoff[3] = {0, 576, 2880};
  for (int i = 0; i < 3; ++i) {
    size_t tot = (size_t)B_ * fHW[i] * fC[i];
    feat2rm_kernel<<<gblk(tot), 256, 0, stream>>>(feats[i], AT, fC[i], fHW[i], (int)tot);
    gemm(AT, pwc[i], pb[i], Pbuf, nullptr, B_ * fHW[i], 256, fC[i], 0);
    groupnorm_kernel<<<dim3(B_ * 32), 256, 0, stream>>>(
        Pbuf, pgs[i], pgb[i], fHW[i], 0, qoff[i], srcf, src_bf, nullptr, nullptr);
  }

  // ---- encoder layers ----
  const int rows = B_ * LEN_;
  for (int l = 0; l < 6; ++l) {
    addpos_kernel<<<gblk(TOK), 256, 0, stream>>>(srcf, pos, q_bf, (int)TOK);
    gemm(src_bf, val_wT + l * 65536, val_b + l * 256, value, nullptr, rows, 256, 256, 0);
    gemm(q_bf, off_wT + l * 49152, off_b + l * 192, offv, nullptr, rows, 192, 256, 0);
    gemm(q_bf, aw_wT + l * 24576, aw_b + l * 96, awv, nullptr, rows, 96, 256, 0);
    softmax12_kernel<<<gblk((size_t)rows * NH_), 256, 0, stream>>>(awv, rows * NH_);
    {
      int waves = rows * NH_;
      msdeform_kernel<<<dim3((waves + 7) / 8), 256, 0, stream>>>(value, offv, awv, samp_bf);
    }
    gemm(samp_bf, out_wT + l * 65536, out_b + l * 256, tmp256, nullptr, rows, 256, 256, 0);
    resln_kernel<<<dim3((rows + 7) / 8), 256, 0, stream>>>(
        srcf, tmp256, ln1_s + l * 256, ln1_b + l * 256, src_bf, rows);
    gemm(src_bf, f1T + l * 262144, ffn1_b + l * 1024, nullptr, hidden, rows, 1024, 256, 1);
    gemm(hidden, f2T + l * 262144, ffn2_b + l * 256, tmp256, nullptr, rows, 256, 1024, 0);
    resln_kernel<<<dim3((rows + 7) / 8), 256, 0, stream>>>(
        srcf, tmp256, ln2_s + l * 256, ln2_b + l * 256, src_bf, rows);
  }

  // ---- multi-scale outputs (CHW) ----
  size_t oofs = (size_t)B_ * 256 * 36864;            // y first
  for (int i = 0; i < 3; ++i) {
    size_t tot = (size_t)B_ * 256 * fHW[i];
    copyout_kernel<<<gblk(tot), 256, 0, stream>>>(srcf, out + oofs, qoff[i], fHW[i], (int)tot);
    oofs += tot;
  }

  // ---- FPN tail ----
  {
    size_t tot = (size_t)B_ * 36864 * 256;
    feat2rm_kernel<<<gblk(tot), 256, 0, stream>>>(res2, AT, 256, 36864, (int)tot);
    gemm(AT, latc, nullptr, Pbuf, nullptr, B_ * 36864, 256, 256, 0);
    groupnorm_kernel<<<dim3(B_ * 32), 256, 0, stream>>>(
        Pbuf, lat_gs, lat_gb, 36864, 1, 0, nullptr, nullptr, latb, nullptr);
    upsample_add_kernel<<<gblk(tot), 256, 0, stream>>>(latb, srcf, AT, (int)tot); // AT = y_bf
    int tiles = (B_ * 36864 / 16) * 4;
    conv3x3_wmma_kernel<<<dim3((tiles + 7) / 8), 256, 0, stream>>>(AT, fpnT, Pbuf, tiles);
    groupnorm_kernel<<<dim3(B_ * 32), 256, 0, stream>>>(
        Pbuf, fpn_gs, fpn_gb, 36864, 2, 0, nullptr, nullptr, nullptr, out);
  }
}
